// SASRec_17660905521565
// MI455X (gfx1250) — matrix operand order repaired and checked by reading, not compiled, not run on hardware
//
#include <hip/hip_runtime.h>
#include <hip/hip_bf16.h>

// ---------------------------------------------------------------------------
// SASRec forward on gfx1250 (MI455X): bf16 WMMA GEMMs (v_wmma_f32_16x16x32_bf16)
// with fp32 accumulation; fp32 softmax / layernorm. Compute-bound (~275 GFLOP,
// only ~170MB unavoidable HBM traffic at 23.3 TB/s). A-tiles staged into LDS
// via the Tensor Data Mover (tensor_load_to_lds + s_wait_tensorcnt) when the
// builtin is available; manual staging otherwise.
// ---------------------------------------------------------------------------

typedef float        v8f   __attribute__((ext_vector_type(8)));
typedef __bf16       v16bf __attribute__((ext_vector_type(16)));
typedef __bf16       v8bf  __attribute__((ext_vector_type(8)));
typedef unsigned int u32x4 __attribute__((ext_vector_type(4)));
typedef int          i32x4 __attribute__((ext_vector_type(4)));
typedef int          i32x8 __attribute__((ext_vector_type(8)));

#if defined(__AMDGCN__) && __has_builtin(__builtin_amdgcn_tensor_load_to_lds) && \
    __has_builtin(__builtin_amdgcn_s_wait_tensorcnt)
#define USE_TDM 1
#else
#define USE_TDM 0
#endif

__device__ __forceinline__ v8f wmma_bf16(v16bf a, v16bf b, v8f c) {
  // (neg_a, A, neg_b, B, c_mod, C, reuse_a, reuse_b)
  return __builtin_amdgcn_wmma_f32_16x16x32_bf16(false, a, false, b, (short)0, c,
                                                 false, false);
}

// ------------------------------- f32 -> bf16 -------------------------------
__global__ void cvt_f32_bf16(const float* __restrict__ src,
                             __bf16* __restrict__ dst, size_t n) {
  size_t i = (size_t)blockIdx.x * blockDim.x + threadIdx.x;
  size_t stride = (size_t)gridDim.x * blockDim.x;
  for (; i < n; i += stride) dst[i] = (__bf16)src[i];
}

// ------------------------------- WMMA GEMM ---------------------------------
// C[M,N] = A[M,K] @ B[K,N] (+bias) (+pos_emb) (ReLU?) -> f32 and/or bf16 out.
// Block tile 128x64, 8 waves (each 32x32 = 2x2 WMMA frags), K step 32.
// A tile (128x32 bf16, LDS row stride 40) staged by the TDM: 2D descriptor,
// data_size=2B, tile=(32,128), tensor_dim0_stride=K elements, pad_enable with
// pad_interval=16 DWORDs (one 64B row) and pad_amount=4 DWORDs (16B) -> the
// 40-element padded LDS stride. All waves issue the same idempotent DMA
// (TDM ignores EXEC), wait on their own TENSORcnt, then barrier.
template <bool RELU, bool ADDPOS, bool OUTF, bool OUTB, bool TRANSB>
__global__ __launch_bounds__(256) void gemm_wmma(
    const __bf16* __restrict__ A, const __bf16* __restrict__ Bm,
    const float* __restrict__ bias, const float* __restrict__ pos,
    float* __restrict__ outF, __bf16* __restrict__ outB, int M, int N, int K,
    int S, int Hn, int DKc) {
  __shared__ __bf16 sA[128][40];   // row-major A tile (+pad)
  __shared__ __bf16 sBT[64][40];   // transposed B tile: [n][k] (+pad)

  const int tid = threadIdx.x;
  const int lane = tid & 31;
  const int wave = tid >> 5;
  const int waveM = wave & 3;   // 4 waves along M (32 rows each)
  const int waveN = wave >> 2;  // 2 waves along N (32 cols each)
  const int lh = lane >> 4;     // half-wave
  const int lm = lane & 15;

  const int m0 = blockIdx.y * 128;
  const int n0 = blockIdx.x * 64;

  // staging assignments
  const int arow = tid >> 1;         // 0..127
  const int acol = (tid & 1) * 16;   // 0 / 16
  const int bkr = tid >> 3;          // 0..31
  const int bnc = (tid & 7) * 8;     // 0..56

  v8f zero = {0.f, 0.f, 0.f, 0.f, 0.f, 0.f, 0.f, 0.f};
  v8f acc[2][2];
#pragma unroll
  for (int i = 0; i < 2; ++i)
#pragma unroll
    for (int j = 0; j < 2; ++j) acc[i][j] = zero;

  for (int k0 = 0; k0 < K; k0 += 32) {
#if USE_TDM
    {
      // ---- Tensor Data Mover: global A tile -> padded LDS tile ----
      unsigned long long ga =
          (unsigned long long)(const void*)&A[(size_t)m0 * K + k0];
      unsigned ldsa = (unsigned)(unsigned long long)(void*)&sA[0][0];
      u32x4 g0;
      g0[0] = 1u;                         // count=1, user descriptor
      g0[1] = ldsa;                       // lds_addr (bytes)
      g0[2] = (unsigned)ga;               // global_addr[31:0]
      g0[3] = (unsigned)((ga >> 32) & 0x01FFFFFFu) | (2u << 30);  // type=2
      i32x8 g1;
      // data_size=2B(code 1), pad_enable, pad_interval=16DW(3), pad_amount=4DW(3)
      g1[0] = (1 << 16) | (1 << 20) | (3 << 22) | (3 << 25);
      unsigned d0 = (unsigned)K;          // tensor_dim0 (elements)
      unsigned d1 = (unsigned)M;          // tensor_dim1 (rows)
      g1[1] = (int)((d0 & 0xFFFFu) << 16);                 // dim0[15:0]
      g1[2] = (int)((d0 >> 16) | ((d1 & 0xFFFFu) << 16));  // dim0[31:16]|dim1[15:0]
      g1[3] = (int)((d1 >> 16) | (32u << 16));             // dim1[31:16]|tile_dim0=32
      g1[4] = 128;                                         // tile_dim1=128, tile_dim2=0
      g1[5] = (int)(unsigned)K;                            // dim0_stride lo (elements)
      g1[6] = 0;
      g1[7] = 0;
      i32x4 gz = {0, 0, 0, 0};
#if __clang_major__ >= 23
      i32x8 gz8 = {0, 0, 0, 0, 0, 0, 0, 0};
      __builtin_amdgcn_tensor_load_to_lds(g0, g1, gz, gz, gz8, 0);
#else
      __builtin_amdgcn_tensor_load_to_lds(g0, g1, gz, gz, 0);
#endif
    }
#else
    // manual A staging (contiguous 32B per thread)
    *(v16bf*)&sA[arow][acol] =
        *(const v16bf*)&A[(size_t)(m0 + arow) * K + k0 + acol];
    if (k0 + 32 < K)
      __builtin_prefetch(&A[(size_t)(m0 + arow) * K + k0 + 32 + acol], 0, 0);
#endif
    // stage B tile transposed (overlaps with TDM)
    v8bf bv = *(const v8bf*)&Bm[(size_t)(k0 + bkr) * N + n0 + bnc];
#pragma unroll
    for (int j = 0; j < 8; ++j) sBT[bnc + j][bkr] = bv[j];
    if (k0 + 32 < K)
      __builtin_prefetch(&Bm[(size_t)(k0 + 32 + bkr) * N + n0 + bnc], 0, 0);
#if USE_TDM
    __builtin_amdgcn_s_wait_tensorcnt(0);
#endif
    __syncthreads();

    // A fragments: lane=row(lm), K = 16*lh + i
    v16bf aF[2];
#pragma unroll
    for (int i = 0; i < 2; ++i) {
      int r = waveM * 32 + i * 16 + lm;
      v8bf lo = *(const v8bf*)&sA[r][16 * lh];
      v8bf hi = *(const v8bf*)&sA[r][16 * lh + 8];
      v16bf t;
#pragma unroll
      for (int e = 0; e < 8; ++e) { t[e] = lo[e]; t[e + 8] = hi[e]; }
      aF[i] = t;
    }
    // B fragments: lane=col(lm), K = 16*lh + i  (contiguous in sBT)
    v16bf bF[2];
#pragma unroll
    for (int j = 0; j < 2; ++j) {
      int n = waveN * 32 + j * 16 + lm;
      bF[j] = *(const v16bf*)&sBT[n][16 * lh];
    }
#pragma unroll
    for (int i = 0; i < 2; ++i)
#pragma unroll
      for (int j = 0; j < 2; ++j) acc[i][j] = wmma_bf16(aF[i], bF[j], acc[i][j]);
    __syncthreads();
  }

  // epilogue: C layout row = base + r + 8*lh, col = base + lm
#pragma unroll
  for (int i = 0; i < 2; ++i)
#pragma unroll
    for (int j = 0; j < 2; ++j) {
      int col = n0 + waveN * 32 + j * 16 + lm;
      float bterm = bias ? bias[col] : 0.f;
      int rbase = m0 + waveM * 32 + i * 16 + 8 * lh;
#pragma unroll
      for (int r = 0; r < 8; ++r) {
        int row = rbase + r;
        float v = acc[i][j][r] + bterm;
        if (ADDPOS) v += pos[(size_t)(row % S) * N + col];
        if (RELU) v = v > 0.f ? v : 0.f;
        if (OUTF) outF[(size_t)row * N + col] = v;
        if (OUTB) {
          __bf16 bvv = (__bf16)v;
          if (TRANSB) {  // store per-head transposed: vT[((b*H+h)*DK+d)*S + s]
            int bb = row / S, s = row % S;
            int hh = col / DKc, d = col % DKc;
            outB[((size_t)(bb * Hn + hh) * DKc + d) * S + s] = bvv;
          } else {
            outB[(size_t)row * N + col] = bvv;
          }
        }
      }
    }
}

// ------------------------------- attention ---------------------------------
// One wave per (b, h, 16-query tile). Faithful-bug masking: masked scores are
// ZERO (not -inf) -> exp(0)=1 terms stay in the softmax, so full SxS P@V.
__global__ __launch_bounds__(32) void attn_kernel(
    const __bf16* __restrict__ q, const __bf16* __restrict__ k,
    const __bf16* __restrict__ vT, __bf16* __restrict__ ctx, int Bn, int Sn,
    int En, int Hn, int DKc) {
  __shared__ float sS[16][512];
  __shared__ __bf16 sP[16][512];
  __shared__ float sZ[16];

  const int QT = Sn / 16;  // 32
  int blk = blockIdx.x;
  int qt = blk % QT;
  int bh = blk / QT;
  int b = bh / Hn, h = bh % Hn;
  int lane = threadIdx.x;
  int lh = lane >> 4, lm = lane & 15;
  size_t rowQ = (size_t)b * Sn + qt * 16 + lm;
  size_t cbase = (size_t)h * DKc;

  v8f zero = {0.f, 0.f, 0.f, 0.f, 0.f, 0.f, 0.f, 0.f};

  // Q fragments straight from global (A layout: two contiguous 16B loads)
  v16bf qF[2];
#pragma unroll
  for (int c = 0; c < 2; ++c) {
    v8bf lo = *(const v8bf*)&q[rowQ * En + cbase + c * 32 + 16 * lh];
    v8bf hi = *(const v8bf*)&q[rowQ * En + cbase + c * 32 + 16 * lh + 8];
    v16bf t;
#pragma unroll
    for (int e = 0; e < 8; ++e) { t[e] = lo[e]; t[e + 8] = hi[e]; }
    qF[c] = t;
  }

  // ---- scores = Q @ K^T, mask*scale, into LDS ----
  for (int kt = 0; kt < QT; ++kt) {
    size_t rowK = (size_t)b * Sn + kt * 16 + lm;  // lane = N = k-row
    v16bf k0 = *(const v16bf*)&k[rowK * En + cbase + 0 + 16 * lh];
    v16bf k1 = *(const v16bf*)&k[rowK * En + cbase + 32 + 16 * lh];
    v8f s = wmma_bf16(qF[0], k0, zero);
    s = wmma_bf16(qF[1], k1, s);
    int colA = kt * 16 + lm;
#pragma unroll
    for (int r = 0; r < 8; ++r) {
      int rowA = qt * 16 + r + 8 * lh;
      sS[r + 8 * lh][colA] = (colA <= rowA) ? s[r] * 0.125f : 0.f;
    }
  }
  __syncthreads();

  // ---- row softmax stats (lane -> row lm, half lh scans 256 cols) ----
  {
    float m = -1e30f;
    for (int c = 0; c < 256; ++c) m = fmaxf(m, sS[lm][lh * 256 + c]);
    m = fmaxf(m, __shfl_xor(m, 16));
    float sum = 0.f;
    for (int c = 0; c < 256; ++c) {
      float e = __expf(sS[lm][lh * 256 + c] - m);
      sum += e;
      sP[lm][lh * 256 + c] = (__bf16)e;  // unnormalized; 1/Z folded later
    }
    sum += __shfl_xor(sum, 16);
    if (lane < 16) sZ[lm] = sum;
  }
  __syncthreads();

  // ---- ctx = P @ V (A from LDS, B contiguous from vT) ----
  v8f acc[4];
#pragma unroll
  for (int dn = 0; dn < 4; ++dn) acc[dn] = zero;
  for (int kc = 0; kc < Sn / 32; ++kc) {
    v8bf lo = *(const v8bf*)&sP[lm][kc * 32 + 16 * lh];
    v8bf hi = *(const v8bf*)&sP[lm][kc * 32 + 16 * lh + 8];
    v16bf a;
#pragma unroll
    for (int e = 0; e < 8; ++e) { a[e] = lo[e]; a[e + 8] = hi[e]; }
#pragma unroll
    for (int dn = 0; dn < 4; ++dn) {
      const __bf16* vp =
          &vT[((size_t)(b * Hn + h) * DKc + dn * 16 + lm) * Sn + kc * 32 +
              16 * lh];
      acc[dn] = wmma_bf16(a, *(const v16bf*)vp, acc[dn]);
    }
  }

  float invZ[8];
#pragma unroll
  for (int r = 0; r < 8; ++r) invZ[r] = 1.f / sZ[r + 8 * lh];
#pragma unroll
  for (int dn = 0; dn < 4; ++dn)
#pragma unroll
    for (int r = 0; r < 8; ++r) {
      int rowL = r + 8 * lh;
      size_t orow = (size_t)b * Sn + qt * 16 + rowL;
      ctx[orow * En + cbase + dn * 16 + lm] = (__bf16)(acc[dn][r] * invZ[r]);
    }
}

// --------------------------- residual + LayerNorm --------------------------
// x = LN(x + res) ; res row is (row % S) when bcast0 (faithful attn[0] bug).
__global__ __launch_bounds__(256) void ln_residual(
    const float* __restrict__ x, const float* __restrict__ res,
    const float* __restrict__ g, const float* __restrict__ bta,
    float* __restrict__ outF, __bf16* __restrict__ outB, int E_, int S_,
    int bcast0) {
  __shared__ float red[256];
  int row = blockIdx.x;
  int tid = threadIdx.x;
  size_t rb = (size_t)row * E_;
  size_t rr = bcast0 ? (size_t)(row % S_) * E_ : rb;
  float v0 = x[rb + tid] + res[rr + tid];
  float v1 = x[rb + tid + 256] + res[rr + tid + 256];

  red[tid] = v0 + v1;
  __syncthreads();
  for (int s = 128; s > 0; s >>= 1) {
    if (tid < s) red[tid] += red[tid + s];
    __syncthreads();
  }
  float mean = red[0] / E_;
  __syncthreads();
  float d0 = v0 - mean, d1 = v1 - mean;
  red[tid] = d0 * d0 + d1 * d1;
  __syncthreads();
  for (int s = 128; s > 0; s >>= 1) {
    if (tid < s) red[tid] += red[tid + s];
    __syncthreads();
  }
  float rstd = rsqrtf(red[0] / E_ + 1e-5f);
  float y0 = d0 * rstd * g[tid] + bta[tid];
  float y1 = d1 * rstd * g[tid + 256] + bta[tid + 256];
  outF[rb + tid] = y0;
  outF[rb + tid + 256] = y1;
  outB[rb + tid] = (__bf16)y0;
  outB[rb + tid + 256] = (__bf16)y1;
}

// ------------------------------- final head --------------------------------
__global__ __launch_bounds__(256) void fc_kernel(const float* __restrict__ x,
                                                 const float* __restrict__ w,
                                                 const float* __restrict__ b,
                                                 float* __restrict__ out,
                                                 int E_) {
  int wave = threadIdx.x >> 5, lane = threadIdx.x & 31;
  int row = blockIdx.x * 8 + wave;
  float s = 0.f;
  for (int c = lane; c < E_; c += 32) s += x[(size_t)row * E_ + c] * w[c];
#pragma unroll
  for (int m = 16; m > 0; m >>= 1) s += __shfl_xor(s, m);
  if (lane == 0) out[row] = s + b[0];
}

// ------------------------------- launcher ----------------------------------
extern "C" void kernel_launch(void* const* d_in, const int* in_sizes, int n_in,
                              void* d_out, int out_size, void* d_ws,
                              size_t ws_size, hipStream_t stream) {
  (void)in_sizes; (void)n_in; (void)out_size; (void)ws_size;
  const int Bc = 32, Sc = 512, NI = 2048, Ec = 512, Hc = 8, Lc = 2,
            DFFc = 2048, DKc = 64;
  const int Mr = Bc * Sc;  // 16384 rows

  const float* input_seq = (const float*)d_in[0];
  const float* emb_W = (const float*)d_in[1];
  const float* emb_b = (const float*)d_in[2];
  const float* pos_emb = (const float*)d_in[3];
  const float* Wq = (const float*)d_in[4];
  const float* bq = (const float*)d_in[5];
  const float* Wk = (const float*)d_in[6];
  const float* bk = (const float*)d_in[7];
  const float* Wv = (const float*)d_in[8];
  const float* bv = (const float*)d_in[9];
  const float* Wo = (const float*)d_in[10];
  const float* bo = (const float*)d_in[11];
  const float* ln1_g = (const float*)d_in[12];
  const float* ln1_b = (const float*)d_in[13];
  const float* W1 = (const float*)d_in[14];
  const float* b1 = (const float*)d_in[15];
  const float* W2 = (const float*)d_in[16];
  const float* b2 = (const float*)d_in[17];
  const float* ln2_g = (const float*)d_in[18];
  const float* ln2_b = (const float*)d_in[19];
  const float* fcW = (const float*)d_in[20];
  const float* fcb = (const float*)d_in[21];

  // ---- workspace carve ----
  char* wsp = (char*)d_ws;
  auto carve = [&](size_t bytes) -> void* {
    void* p = (void*)wsp;
    wsp += (bytes + 255) & ~(size_t)255;
    return p;
  };
  __bf16* inb = (__bf16*)carve((size_t)Mr * NI * 2);  // reused as FFN hidden
  __bf16* wEmb = (__bf16*)carve((size_t)NI * Ec * 2);
  __bf16 *wQb[Lc], *wKb[Lc], *wVb[Lc], *wOb[Lc], *wF1b[Lc], *wF2b[Lc];
  for (int l = 0; l < Lc; ++l) {
    wQb[l] = (__bf16*)carve((size_t)Ec * Ec * 2);
    wKb[l] = (__bf16*)carve((size_t)Ec * Ec * 2);
    wVb[l] = (__bf16*)carve((size_t)Ec * Ec * 2);
    wOb[l] = (__bf16*)carve((size_t)Ec * Ec * 2);
    wF1b[l] = (__bf16*)carve((size_t)Ec * DFFc * 2);
    wF2b[l] = (__bf16*)carve((size_t)DFFc * Ec * 2);
  }
  float* xf = (float*)carve((size_t)Mr * Ec * 4);
  __bf16* xb = (__bf16*)carve((size_t)Mr * Ec * 2);
  __bf16* qb = (__bf16*)carve((size_t)Mr * Ec * 2);
  __bf16* kb = (__bf16*)carve((size_t)Mr * Ec * 2);
  __bf16* vTb = (__bf16*)carve((size_t)Mr * Ec * 2);  // [B*H*DK][S]
  __bf16* ctxb = (__bf16*)carve((size_t)Mr * Ec * 2);
  float* projf = (float*)carve((size_t)Mr * Ec * 4);
  __bf16* hb = inb;  // FFN hidden reuses input buffer (done after embedding)

  // ---- convert inputs/weights to bf16 ----
  cvt_f32_bf16<<<4096, 256, 0, stream>>>(input_seq, inb, (size_t)Mr * NI);
  cvt_f32_bf16<<<512, 256, 0, stream>>>(emb_W, wEmb, (size_t)NI * Ec);
  for (int l = 0; l < Lc; ++l) {
    cvt_f32_bf16<<<256, 256, 0, stream>>>(Wq + (size_t)l * Ec * Ec, wQb[l],
                                          (size_t)Ec * Ec);
    cvt_f32_bf16<<<256, 256, 0, stream>>>(Wk + (size_t)l * Ec * Ec, wKb[l],
                                          (size_t)Ec * Ec);
    cvt_f32_bf16<<<256, 256, 0, stream>>>(Wv + (size_t)l * Ec * Ec, wVb[l],
                                          (size_t)Ec * Ec);
    cvt_f32_bf16<<<256, 256, 0, stream>>>(Wo + (size_t)l * Ec * Ec, wOb[l],
                                          (size_t)Ec * Ec);
    cvt_f32_bf16<<<512, 256, 0, stream>>>(W1 + (size_t)l * Ec * DFFc, wF1b[l],
                                          (size_t)Ec * DFFc);
    cvt_f32_bf16<<<512, 256, 0, stream>>>(W2 + (size_t)l * DFFc * Ec, wF2b[l],
                                          (size_t)DFFc * Ec);
  }

  dim3 blk(256);
  // embedding: x = input @ emb_W + emb_b + pos_emb  (f32 + bf16 copies)
  gemm_wmma<false, true, true, true, false>
      <<<dim3(Ec / 64, Mr / 128), blk, 0, stream>>>(
          inb, wEmb, emb_b, pos_emb, xf, xb, Mr, Ec, NI, Sc, Hc, DKc);

  for (int l = 0; l < Lc; ++l) {
    // Q, K (row layout), V (head-transposed layout)
    gemm_wmma<false, false, false, true, false>
        <<<dim3(Ec / 64, Mr / 128), blk, 0, stream>>>(
            xb, wQb[l], bq + l * Ec, nullptr, nullptr, qb, Mr, Ec, Ec, Sc, Hc,
            DKc);
    gemm_wmma<false, false, false, true, false>
        <<<dim3(Ec / 64, Mr / 128), blk, 0, stream>>>(
            xb, wKb[l], bk + l * Ec, nullptr, nullptr, kb, Mr, Ec, Ec, Sc, Hc,
            DKc);
    gemm_wmma<false, false, false, true, true>
        <<<dim3(Ec / 64, Mr / 128), blk, 0, stream>>>(
            xb, wVb[l], bv + l * Ec, nullptr, nullptr, vTb, Mr, Ec, Ec, Sc, Hc,
            DKc);
    // attention (masked-to-zero softmax, full SxS P@V)
    attn_kernel<<<Bc * Hc * (Sc / 16), 32, 0, stream>>>(qb, kb, vTb, ctxb, Bc,
                                                        Sc, Ec, Hc, DKc);
    // output projection
    gemm_wmma<false, false, true, false, false>
        <<<dim3(Ec / 64, Mr / 128), blk, 0, stream>>>(
            ctxb, wOb[l], bo + l * Ec, nullptr, projf, nullptr, Mr, Ec, Ec, Sc,
            Hc, DKc);
    // x = LN(x + proj[batch0 broadcast])
    ln_residual<<<Mr, 256, 0, stream>>>(xf, projf, ln1_g + l * Ec,
                                        ln1_b + l * Ec, xf, xb, Ec, Sc, 1);
    // FFN
    gemm_wmma<true, false, false, true, false>
        <<<dim3(DFFc / 64, Mr / 128), blk, 0, stream>>>(
            xb, wF1b[l], b1 + l * DFFc, nullptr, nullptr, hb, Mr, DFFc, Ec, Sc,
            Hc, DKc);
    gemm_wmma<false, false, true, false, false>
        <<<dim3(Ec / 64, Mr / 128), blk, 0, stream>>>(
            hb, wF2b[l], b2 + l * Ec, nullptr, projf, nullptr, Mr, Ec, DFFc,
            Sc, Hc, DKc);
    ln_residual<<<Mr, 256, 0, stream>>>(xf, projf, ln2_g + l * Ec,
                                        ln2_b + l * Ec, xf, xb, Ec, Sc, 0);
  }

  fc_kernel<<<Mr / 8, 256, 0, stream>>>(xf, fcW, fcb, (float*)d_out, Ec);
}
